// MultimodalLongFormer_26723286516192
// MI455X (gfx1250) — compile-verified, matrix operand order checked by reading, not run on hardware
//
#include <hip/hip_runtime.h>
#include <math.h>

typedef __bf16 bf16;
typedef __attribute__((ext_vector_type(16))) __bf16 v16bf;
typedef __attribute__((ext_vector_type(8)))  __bf16 bf16x8;
typedef __attribute__((ext_vector_type(8)))  float   v8f;

#define DEV __device__ __forceinline__

#if defined(__has_builtin)
#if __has_builtin(__builtin_amdgcn_global_load_async_to_lds_b128) && \
    __has_builtin(__builtin_amdgcn_s_wait_asynccnt)
#define USE_ASYNC_LDS 1
#endif
#endif

#ifdef USE_ASYNC_LDS
#define AS1 __attribute__((address_space(1)))
#define AS3 __attribute__((address_space(3)))
typedef int v4i __attribute__((vector_size(16)));
#endif

DEV bf16 f2b(float f) {
  unsigned u = __builtin_bit_cast(unsigned, f);
  unsigned r = (u + 0x7FFFu + ((u >> 16) & 1u)) >> 16;
  unsigned short s = (unsigned short)r;
  return __builtin_bit_cast(bf16, s);
}

DEV float gelu_exact(float x) {
  return 0.5f * x * (1.0f + erff(x * 0.70710678118654752f));
}

// ---------------------------------------------------------------------------
// f32 -> bf16 bulk convert (4-wide)
// ---------------------------------------------------------------------------
__global__ __launch_bounds__(256)
void cvt_kernel(const float* __restrict__ src, bf16* __restrict__ dst, size_t n4) {
  size_t i = (size_t)blockIdx.x * blockDim.x + threadIdx.x;
  size_t stride = (size_t)gridDim.x * blockDim.x;
  for (; i < n4; i += stride) {
    float4 v = ((const float4*)src)[i];
    ushort4 p;
    p.x = __builtin_bit_cast(unsigned short, f2b(v.x));
    p.y = __builtin_bit_cast(unsigned short, f2b(v.y));
    p.z = __builtin_bit_cast(unsigned short, f2b(v.z));
    p.w = __builtin_bit_cast(unsigned short, f2b(v.w));
    ((ushort4*)dst)[i] = p;
  }
}

// ---------------------------------------------------------------------------
// WMMA GEMM: C(M,N) = A(M,K)bf16 * W(N,K)bf16^T + bias, f32 accumulate.
// Block tile 128x128 (8 waves, wave tile 32x64). K step 32.
// Double-buffered LDS; tile k+1 is copied while the WMMAs of tile k run.
// GLOBAL_LOAD_ASYNC_TO_LDS_B128 moves the tile straight into LDS on the
// async engine (ASYNCcnt); fallback stages through VGPRs.
// ---------------------------------------------------------------------------
#define BM 128
#define BN 128
#define BK 32
#define LDT 40   // LDS row stride (bf16) -> 80B rows, 16B aligned fragments

DEV v16bf frag_load(const bf16* base, int row0, int lane) {
  // 16-bit A/B-operand layout: lanes 0-15 hold row, K 0-7 (vgpr0-3) and
  // 16-23 (vgpr4-7); lanes 16-31 hold K 8-15 and 24-31.
  int r  = row0 + (lane & 15);
  int kh = (lane >> 4) << 3;
  const bf16* p = base + r * LDT + kh;
  bf16x8 lo = *(const bf16x8*)(p);
  bf16x8 hi = *(const bf16x8*)(p + 16);
  return __builtin_shufflevector(lo, hi, 0,1,2,3,4,5,6,7,8,9,10,11,12,13,14,15);
}

#ifdef USE_ASYNC_LDS
DEV void async_tile(const bf16* __restrict__ A, const bf16* __restrict__ W,
                    bf16* As, bf16* Bs, int m0, int n0, int K, int k0, int tid) {
  #pragma unroll
  for (int it = 0; it < 2; ++it) {
    int idx = (tid + it * 256) * 8;   // 4096 bf16 per tile
    int r = idx >> 5;
    int c = idx & 31;
    __builtin_amdgcn_global_load_async_to_lds_b128(
        (AS1 v4i*)(A + (size_t)(m0 + r) * K + (k0 + c)),
        (AS3 v4i*)(&As[r * LDT + c]), 0, 0);
    __builtin_amdgcn_global_load_async_to_lds_b128(
        (AS1 v4i*)(W + (size_t)(n0 + r) * K + (k0 + c)),
        (AS3 v4i*)(&Bs[r * LDT + c]), 0, 0);
  }
}
#else
DEV void issue_loads(const bf16* __restrict__ A, const bf16* __restrict__ W,
                     int m0, int n0, int K, int k0, int tid,
                     bf16x8* ra, bf16x8* rb) {
  #pragma unroll
  for (int it = 0; it < 2; ++it) {
    int idx = (tid + it * 256) * 8;   // 4096 bf16 per tile
    int r = idx >> 5;
    int c = idx & 31;
    ra[it] = *(const bf16x8*)(A + (size_t)(m0 + r) * K + (k0 + c));
    rb[it] = *(const bf16x8*)(W + (size_t)(n0 + r) * K + (k0 + c));
  }
}

DEV void store_tile(bf16* As, bf16* Bs, int tid, const bf16x8* ra, const bf16x8* rb) {
  #pragma unroll
  for (int it = 0; it < 2; ++it) {
    int idx = (tid + it * 256) * 8;
    int r = idx >> 5;
    int c = idx & 31;
    *(bf16x8*)(&As[r * LDT + c]) = ra[it];
    *(bf16x8*)(&Bs[r * LDT + c]) = rb[it];
  }
}
#endif

template<int ACT_GELU>
__global__ __launch_bounds__(256)
void gemm_kernel(const bf16* __restrict__ A, const bf16* __restrict__ W,
                 const float* __restrict__ bias,
                 float* Cf, bf16* Cb, int M, int N, int K) {
  __shared__ bf16 As[2][BM * LDT];
  __shared__ bf16 Bs[2][BN * LDT];
  const int tid  = threadIdx.x;
  const int lane = tid & 31;
  const int wid  = tid >> 5;
  const int m0 = blockIdx.y * BM;
  const int n0 = blockIdx.x * BN;
  const int wm = (wid >> 1) * 32;   // wave row offset in block tile
  const int wn = (wid & 1) * 64;    // wave col offset in block tile

  v8f acc[2][4];
  #pragma unroll
  for (int i = 0; i < 2; ++i)
    #pragma unroll
    for (int j = 0; j < 4; ++j)
      #pragma unroll
      for (int r = 0; r < 8; ++r) acc[i][j][r] = 0.0f;

  const int nk = K / BK;
  int cur = 0;

#ifdef USE_ASYNC_LDS
  async_tile(A, W, As[0], Bs[0], m0, n0, K, 0, tid);
  __builtin_amdgcn_s_wait_asynccnt(0);
  __syncthreads();
#else
  bf16x8 ra[2], rb[2];
  issue_loads(A, W, m0, n0, K, 0, tid, ra, rb);
  store_tile(As[0], Bs[0], tid, ra, rb);
#endif

  for (int kt = 0; kt < nk; ++kt) {
    const bool has_next = (kt + 1 < nk);
#ifdef USE_ASYNC_LDS
    if (has_next)
      async_tile(A, W, As[cur ^ 1], Bs[cur ^ 1], m0, n0, K, (kt + 1) * BK, tid);
#else
    if (has_next)
      issue_loads(A, W, m0, n0, K, (kt + 1) * BK, tid, ra, rb);
#endif
    if (kt + 2 < nk) {  // global_prefetch_b8 two tiles ahead
      __builtin_prefetch(A + (size_t)(m0 + (tid >> 1)) * K + ((kt + 2) * BK), 0, 2);
      __builtin_prefetch(W + (size_t)(n0 + (tid >> 1)) * K + ((kt + 2) * BK), 0, 2);
    }
#ifndef USE_ASYNC_LDS
    __syncthreads();  // previous iteration's LDS stores now visible
#endif

    v16bf af[2], bfr[4];
    #pragma unroll
    for (int i = 0; i < 2; ++i) af[i]  = frag_load(As[cur], wm + i * 16, lane);
    #pragma unroll
    for (int j = 0; j < 4; ++j) bfr[j] = frag_load(Bs[cur], wn + j * 16, lane);
    #pragma unroll
    for (int i = 0; i < 2; ++i)
      #pragma unroll
      for (int j = 0; j < 4; ++j)
        acc[i][j] = __builtin_amdgcn_wmma_f32_16x16x32_bf16(
            false, af[i], false, bfr[j], (short)0, acc[i][j], false, false);

#ifdef USE_ASYNC_LDS
    if (has_next) __builtin_amdgcn_s_wait_asynccnt(0);
    __syncthreads();  // publish tile kt+1, and all reads of buf[cur] done
#else
    if (has_next)
      store_tile(As[cur ^ 1], Bs[cur ^ 1], tid, ra, rb);
#endif
    cur ^= 1;
  }

  // Epilogue: C layout = VGPR r holds M = r (lanes 0-15) / r+8 (lanes 16-31)
  #pragma unroll
  for (int i = 0; i < 2; ++i) {
    #pragma unroll
    for (int j = 0; j < 4; ++j) {
      int col   = n0 + wn + j * 16 + (lane & 15);
      int mbase = m0 + wm + i * 16 + ((lane >> 4) << 3);
      float bv = bias[col];
      #pragma unroll
      for (int r = 0; r < 8; ++r) {
        float v = acc[i][j][r] + bv;
        if (ACT_GELU) v = gelu_exact(v);
        size_t o = (size_t)(mbase + r) * N + col;
        if (Cf) Cf[o] = v;
        if (Cb) Cb[o] = f2b(v);
      }
    }
  }
}

// ---------------------------------------------------------------------------
// Windowed attention (w = 5): one wave per (b, h, s); lane owns 2 head dims.
// ---------------------------------------------------------------------------
__global__ __launch_bounds__(256)
void attn_kernel(const float* __restrict__ q, const float* __restrict__ k,
                 const float* __restrict__ v,
                 const int* __restrict__ mask, const int* __restrict__ gmask,
                 bf16* __restrict__ ctxb, int B, int S, int H) {
  const int lane = threadIdx.x & 31;
  int gw = (int)(((size_t)blockIdx.x * blockDim.x + threadIdx.x) >> 5);
  int total = B * H * S;
  if (gw >= total) return;
  int s = gw % S; int t = gw / S; int h = t % H; int b = t / H;
  const float NEG = -3.4028234663852886e38f;
  const int D = H * 64;
  size_t rowq = (size_t)(b * S + s) * D + h * 64 + lane * 2;
  float2 q2 = *(const float2*)(q + rowq);

  float sc[5];
  #pragma unroll
  for (int j = 0; j < 5; ++j) {
    int sp = s + j - 2;
    if (sp < 0 || sp >= S) { sc[j] = NEG; continue; }
    int mi = b * S + sp;
    float am  = (float)(mask[mi] * (gmask[mi] + 1));
    float ext = (1.0f - am) * NEG;
    size_t rk = (size_t)(b * S + sp) * D + h * 64 + lane * 2;
    float2 k2 = *(const float2*)(k + rk);
    float part = q2.x * k2.x + q2.y * k2.y;
    #pragma unroll
    for (int off = 16; off; off >>= 1) part += __shfl_xor(part, off, 32);
    sc[j] = part * 0.125f + ext;   // scale = 1/sqrt(64)
  }
  float m = sc[0];
  #pragma unroll
  for (int j = 1; j < 5; ++j) m = fmaxf(m, sc[j]);
  float p[5]; float sum = 0.0f;
  #pragma unroll
  for (int j = 0; j < 5; ++j) { p[j] = __expf(sc[j] - m); sum += p[j]; }
  float inv = 1.0f / sum;

  int mi0 = b * S + s;
  float amq = (float)(mask[mi0] * (gmask[mi0] + 1));
  bool qmasked = ((1.0f - amq) * NEG) < 0.0f;

  float cx = 0.0f, cy = 0.0f;
  #pragma unroll
  for (int j = 0; j < 5; ++j) {
    int sp = s + j - 2;
    if (sp < 0 || sp >= S) continue;
    float pj = p[j] * inv;
    size_t rv = (size_t)(b * S + sp) * D + h * 64 + lane * 2;
    float2 v2 = *(const float2*)(v + rv);
    cx += pj * v2.x; cy += pj * v2.y;
  }
  if (qmasked) { cx = 0.0f; cy = 0.0f; }
  unsigned lo = __builtin_bit_cast(unsigned short, f2b(cx));
  unsigned hi = __builtin_bit_cast(unsigned short, f2b(cy));
  *(unsigned*)(ctxb + rowq) = lo | (hi << 16);
}

// ---------------------------------------------------------------------------
// LayerNorm (D=768, block=256, 3 elems/thread). Handles:
//   coarse != null : x += coarse_emb[coarse[row]] + pos_emb[row % S]
//   res    != null : x += res[row]
// Emits f32 and optional bf16 copy. res/outf may alias h (block-local rows).
// ---------------------------------------------------------------------------
__global__ __launch_bounds__(256)
void ln_kernel(const float* __restrict__ x, const float* res,
               const int* __restrict__ coarse,
               const float* __restrict__ cemb, const float* __restrict__ pemb,
               const float* __restrict__ g, const float* __restrict__ bta,
               float* outf, bf16* outb, int D, int S) {
  const int row = blockIdx.x;
  const int s = row % S;
  const int tid = threadIdx.x;
  const int lane = tid & 31, wid = tid >> 5;
  const size_t base = (size_t)row * D;
  float vals[3];
  float sum = 0.0f, sq = 0.0f;
  #pragma unroll
  for (int i = 0; i < 3; ++i) {
    int c = tid + i * 256;
    float vv = x[base + c];
    if (res)    vv += res[base + c];
    if (coarse) vv += cemb[(size_t)coarse[row] * D + c] + pemb[(size_t)s * D + c];
    vals[i] = vv; sum += vv; sq += vv * vv;
  }
  #pragma unroll
  for (int off = 16; off; off >>= 1) {
    sum += __shfl_xor(sum, off, 32);
    sq  += __shfl_xor(sq,  off, 32);
  }
  __shared__ float s_s[8], s_q[8];
  if (lane == 0) { s_s[wid] = sum; s_q[wid] = sq; }
  __syncthreads();
  float ts = 0.0f, tq = 0.0f;
  #pragma unroll
  for (int i = 0; i < 8; ++i) { ts += s_s[i]; tq += s_q[i]; }
  float mean = ts / (float)D;
  float var  = tq / (float)D - mean * mean;
  float rstd = rsqrtf(var + 1e-12f);
  #pragma unroll
  for (int i = 0; i < 3; ++i) {
    int c = tid + i * 256;
    float o = (vals[i] - mean) * rstd * g[c] + bta[c];
    outf[base + c] = o;
    if (outb) outb[base + c] = f2b(o);
  }
}

// ---------------------------------------------------------------------------
// Decoder head: out[b, s] = dot(x[b, s, :], dec_w) + dec_b, dropping s = S-1.
// One wave per output row.
// ---------------------------------------------------------------------------
__global__ __launch_bounds__(256)
void dec_kernel(const float* __restrict__ x, const float* __restrict__ w,
                const float* __restrict__ bsc, float* __restrict__ out,
                int B, int S, int D) {
  const int lane = threadIdx.x & 31;
  int gw = (int)(((size_t)blockIdx.x * blockDim.x + threadIdx.x) >> 5);
  int rows = B * (S - 1);
  if (gw >= rows) return;
  int b = gw / (S - 1); int s = gw % (S - 1);
  const float* xp = x + (size_t)(b * S + s) * D;
  float part = 0.0f;
  for (int c = lane; c < D; c += 32) part += xp[c] * w[c];
  #pragma unroll
  for (int off = 16; off; off >>= 1) part += __shfl_xor(part, off, 32);
  if (lane == 0) out[gw] = part + bsc[0];
}

// ---------------------------------------------------------------------------
extern "C" void kernel_launch(void* const* d_in, const int* in_sizes, int n_in,
                              void* d_out, int out_size, void* d_ws, size_t ws_size,
                              hipStream_t stream) {
  (void)in_sizes; (void)n_in; (void)out_size; (void)ws_size;
  const int L = 12, D = 768, H = 12, FF = 3072, Bn = 2, S = 3072, IN = 3584;
  const int M = Bn * S;  // 6144 rows

  const float* input_feat = (const float*)d_in[0];
  const int*   coarse     = (const int*)  d_in[1];
  const int*   mask       = (const int*)  d_in[2];
  const int*   gmask      = (const int*)  d_in[3];
  const float* feat_w     = (const float*)d_in[4];
  const float* feat_b     = (const float*)d_in[5];
  const float* coarse_emb = (const float*)d_in[6];
  const float* pos_emb    = (const float*)d_in[7];
  const float* emb_g      = (const float*)d_in[8];
  const float* emb_b      = (const float*)d_in[9];
  const float* q_w  = (const float*)d_in[10];
  const float* q_b  = (const float*)d_in[11];
  const float* k_w  = (const float*)d_in[12];
  const float* k_b  = (const float*)d_in[13];
  const float* v_w  = (const float*)d_in[14];
  const float* v_b  = (const float*)d_in[15];
  const float* o_w  = (const float*)d_in[16];
  const float* o_b  = (const float*)d_in[17];
  const float* ln1_g = (const float*)d_in[18];
  const float* ln1_b = (const float*)d_in[19];
  const float* i_w  = (const float*)d_in[20];
  const float* i_b  = (const float*)d_in[21];
  const float* f_w  = (const float*)d_in[22];
  const float* f_b  = (const float*)d_in[23];
  const float* ln2_g = (const float*)d_in[24];
  const float* ln2_b = (const float*)d_in[25];
  const float* dense_w = (const float*)d_in[26];
  const float* dense_b = (const float*)d_in[27];
  const float* out_g   = (const float*)d_in[28];
  const float* out_b   = (const float*)d_in[29];
  const float* dec_w   = (const float*)d_in[30];
  const float* dec_b   = (const float*)d_in[31];

  // ---- workspace carve ----
  char* ws = (char*)d_ws;
  size_t off = 0;
  auto alloc = [&](size_t bytes) -> char* {
    char* p = ws + off;
    off = (off + bytes + 255) & ~(size_t)255;
    return p;
  };
  bf16* ifb  = (bf16*)alloc((size_t)M * IN * 2);
  bf16* fwb  = (bf16*)alloc((size_t)D * IN * 2);
  bf16* qwb  = (bf16*)alloc((size_t)L * D * D * 2);
  bf16* kwb  = (bf16*)alloc((size_t)L * D * D * 2);
  bf16* vwb  = (bf16*)alloc((size_t)L * D * D * 2);
  bf16* owb  = (bf16*)alloc((size_t)L * D * D * 2);
  bf16* iwb  = (bf16*)alloc((size_t)L * FF * D * 2);
  bf16* fwb2 = (bf16*)alloc((size_t)L * D * FF * 2);
  bf16* dwb  = (bf16*)alloc((size_t)D * D * 2);
  float* h    = (float*)alloc((size_t)M * D * 4);
  bf16*  hb   = (bf16*) alloc((size_t)M * D * 2);
  float* qf   = (float*)alloc((size_t)M * D * 4);
  float* kf   = (float*)alloc((size_t)M * D * 4);
  float* vf   = (float*)alloc((size_t)M * D * 4);
  bf16*  cxb  = (bf16*) alloc((size_t)M * D * 2);
  float* tmp1 = (float*)alloc((size_t)M * D * 4);
  bf16*  ffb  = (bf16*) alloc((size_t)M * FF * 2);
  float* xout = (float*)alloc((size_t)M * D * 4);

  auto cvt = [&](const float* s, bf16* d, size_t n) {
    size_t n4 = n / 4;
    int blocks = (int)((n4 + 255) / 256);
    if (blocks > 4096) blocks = 4096;
    cvt_kernel<<<blocks, 256, 0, stream>>>(s, d, n4);
  };
  auto gemm = [&](const bf16* A, const bf16* Wt, const float* bias,
                  float* Cf, bf16* Cb, int Mm, int Nn, int Kk, bool act) {
    dim3 grid(Nn / BN, Mm / BM);
    if (act) gemm_kernel<1><<<grid, 256, 0, stream>>>(A, Wt, bias, Cf, Cb, Mm, Nn, Kk);
    else     gemm_kernel<0><<<grid, 256, 0, stream>>>(A, Wt, bias, Cf, Cb, Mm, Nn, Kk);
  };

  // ---- one-time (per launch) bf16 conversions ----
  cvt(input_feat, ifb, (size_t)M * IN);
  cvt(feat_w,     fwb, (size_t)D * IN);
  cvt(q_w, qwb, (size_t)L * D * D);
  cvt(k_w, kwb, (size_t)L * D * D);
  cvt(v_w, vwb, (size_t)L * D * D);
  cvt(o_w, owb, (size_t)L * D * D);
  cvt(i_w, iwb, (size_t)L * FF * D);
  cvt(f_w, fwb2, (size_t)L * D * FF);
  cvt(dense_w, dwb, (size_t)D * D);

  // ---- embedding: h = LN(input @ feat_w^T + feat_b + coarse_emb + pos_emb) ----
  gemm(ifb, fwb, feat_b, tmp1, nullptr, M, D, IN, false);
  ln_kernel<<<M, 256, 0, stream>>>(tmp1, nullptr, coarse, coarse_emb, pos_emb,
                                   emb_g, emb_b, h, hb, D, S);

  // ---- transformer layers ----
  const int attnWaves  = Bn * H * S;
  const int attnBlocks = (attnWaves * 32 + 255) / 256;
  for (int l = 0; l < L; ++l) {
    const size_t wo = (size_t)l * D * D;
    gemm(hb, qwb + wo, q_b + (size_t)l * D, qf, nullptr, M, D, D, false);
    gemm(hb, kwb + wo, k_b + (size_t)l * D, kf, nullptr, M, D, D, false);
    gemm(hb, vwb + wo, v_b + (size_t)l * D, vf, nullptr, M, D, D, false);
    attn_kernel<<<attnBlocks, 256, 0, stream>>>(qf, kf, vf, mask, gmask, cxb, Bn, S, H);
    gemm(cxb, owb + wo, o_b + (size_t)l * D, tmp1, nullptr, M, D, D, false);
    ln_kernel<<<M, 256, 0, stream>>>(tmp1, h, nullptr, nullptr, nullptr,
                                     ln1_g + (size_t)l * D, ln1_b + (size_t)l * D,
                                     h, hb, D, S);
    gemm(hb, iwb + (size_t)l * FF * D, i_b + (size_t)l * FF, nullptr, ffb, M, FF, D, true);
    gemm(ffb, fwb2 + (size_t)l * D * FF, f_b + (size_t)l * D, tmp1, nullptr, M, D, FF, false);
    ln_kernel<<<M, 256, 0, stream>>>(tmp1, h, nullptr, nullptr, nullptr,
                                     ln2_g + (size_t)l * D, ln2_b + (size_t)l * D,
                                     h, hb, D, S);
  }

  // ---- head: gelu(dense) -> LN -> dec dot, drop last position ----
  gemm(hb, dwb, dense_b, tmp1, nullptr, M, D, D, true);
  ln_kernel<<<M, 256, 0, stream>>>(tmp1, nullptr, nullptr, nullptr, nullptr,
                                   out_g, out_b, xout, nullptr, D, S);
  const int rows = Bn * (S - 1);
  const int dblocks = (rows * 32 + 255) / 256;
  dec_kernel<<<dblocks, 256, 0, stream>>>(xout, dec_w, dec_b, (float*)d_out, Bn, S, D);
}
